// PreGatingContextualAttentionGate_72267119722689
// MI455X (gfx1250) — matrix operand order repaired and checked by reading, compiled, not run on hardware
//
#include <hip/hip_runtime.h>
#include <math.h>

typedef __bf16 bf16;
typedef __attribute__((ext_vector_type(16))) __bf16 v16bf;
typedef __attribute__((ext_vector_type(8)))  __bf16 v8bf;
typedef __attribute__((ext_vector_type(8)))  float   v8f;

// ---------------------------------------------------------------------------
// WMMA helper: D = A(16x32 bf16) * B(32x16 bf16) + C(16x16 f32), wave32
// ---------------------------------------------------------------------------
static __device__ __forceinline__ v8f wmma_bf16(v16bf a, v16bf b, v8f c) {
  return __builtin_amdgcn_wmma_f32_16x16x32_bf16(false, a, false, b, (short)0, c,
                                                 false, false);
}

// A fragment (16 rows x 32 K), row-major bf16 source.
// lane m = lane&15, half = lane>>4: elems [k+half*8, +8) and [k+16+half*8, +8)
static __device__ __forceinline__ v16bf load_a_bf16(const bf16* __restrict__ A,
                                                    int lda, int row, int kb,
                                                    int lane) {
  const int m = lane & 15, half = lane >> 4;
  const bf16* p = A + (size_t)(row + m) * lda + kb + half * 8;
  v16bf a;
  *(v8bf*)&a       = *(const v8bf*)p;
  *((v8bf*)&a + 1) = *(const v8bf*)(p + 16);
  return a;
}

// Same A fragment but from f32 source, converted in-register.
static __device__ __forceinline__ v16bf load_a_f32(const float* __restrict__ A,
                                                   int lda, int row, int kb,
                                                   int lane) {
  const int m = lane & 15, half = lane >> 4;
  const float* p = A + (size_t)(row + m) * lda + kb + half * 8;
  const float4 f0 = *(const float4*)(p);
  const float4 f1 = *(const float4*)(p + 4);
  const float4 f2 = *(const float4*)(p + 16);
  const float4 f3 = *(const float4*)(p + 20);
  v16bf a;
  a[0]=(__bf16)f0.x; a[1]=(__bf16)f0.y; a[2]=(__bf16)f0.z; a[3]=(__bf16)f0.w;
  a[4]=(__bf16)f1.x; a[5]=(__bf16)f1.y; a[6]=(__bf16)f1.z; a[7]=(__bf16)f1.w;
  a[8]=(__bf16)f2.x; a[9]=(__bf16)f2.y; a[10]=(__bf16)f2.z; a[11]=(__bf16)f2.w;
  a[12]=(__bf16)f3.x; a[13]=(__bf16)f3.y; a[14]=(__bf16)f3.z; a[15]=(__bf16)f3.w;
  return a;
}

// B fragment (32 K x 16 cols) from "Bt" storage: Bt[n][k] row-major (weight
// style, i.e. C = A * Bt^T).  lane n = lane&15, half = lane>>4:
// elems Bt[n][k+half*8 .. +8) and Bt[n][k+16+half*8 .. +8).
static __device__ __forceinline__ v16bf load_b_bf16(const bf16* __restrict__ Bt,
                                                    int ldb, int col, int kb,
                                                    int lane) {
  const int n = lane & 15, half = lane >> 4;
  const bf16* p = Bt + (size_t)(col + n) * ldb + kb + half * 8;
  v16bf b;
  *(v8bf*)&b       = *(const v8bf*)p;
  *((v8bf*)&b + 1) = *(const v8bf*)(p + 16);
  return b;
}

// ---------------------------------------------------------------------------
// Generic bf16 WMMA GEMM: C[M,N] = A[M,K] @ Bt[N,K]^T (+bias, relu, ...)
// Block: 256 thr = 8 waves as 2(M)x4(N); each wave 32x64 -> block 64x256.
// All epilogue/source options are compile-time to keep the K-loop branch-free.
// ---------------------------------------------------------------------------
template <bool AF32, bool HAS_BIAS, bool RELU, bool OUTF, bool OUTB,
          bool OUTTANH, bool OUTT, bool HASADD>
__global__ __launch_bounds__(256) void k_gemm(
    const bf16* __restrict__ Ab, const float* __restrict__ Af, int lda,
    const bf16* __restrict__ Bt, int ldb, long long bt_bstride, int rpb,
    const float* __restrict__ bias, int M, int N, int K,
    float* __restrict__ outF, bf16* __restrict__ outB,
    bf16* __restrict__ outTanh, bf16* __restrict__ outT,
    const float* __restrict__ addF) {
  const int lane = threadIdx.x & 31;
  const int wave = threadIdx.x >> 5;
  const int wm = wave >> 2, wn = wave & 3;
  const int row0 = blockIdx.y * 64 + wm * 32;
  const int col0 = blockIdx.x * 256 + wn * 64;
  if (row0 >= M || col0 >= N) return;  // wave-uniform (grids are exact)

  const bf16* Bb = Bt;
  if (bt_bstride) Bb += (long long)(row0 / rpb) * bt_bstride;

  const v8f vz = {};
  v8f acc[2][4];
#pragma unroll
  for (int i = 0; i < 2; ++i)
#pragma unroll
    for (int j = 0; j < 4; ++j) acc[i][j] = vz;

  for (int kb = 0; kb < K; kb += 32) {
    // speculative prefetch of the next B k-slice (silently dropped past end)
    __builtin_prefetch(Bb + (size_t)(col0 + (lane & 15)) * ldb + kb + 32, 0, 3);
    v16bf a0, a1;
    if constexpr (AF32) {
      a0 = load_a_f32(Af, lda, row0, kb, lane);
      a1 = load_a_f32(Af, lda, row0 + 16, kb, lane);
    } else {
      a0 = load_a_bf16(Ab, lda, row0, kb, lane);
      a1 = load_a_bf16(Ab, lda, row0 + 16, kb, lane);
    }
    v16bf b[4];
#pragma unroll
    for (int j = 0; j < 4; ++j) b[j] = load_b_bf16(Bb, ldb, col0 + 16 * j, kb, lane);
#pragma unroll
    for (int j = 0; j < 4; ++j) {
      acc[0][j] = wmma_bf16(a0, b[j], acc[0][j]);
      acc[1][j] = wmma_bf16(a1, b[j], acc[1][j]);
    }
  }

  const int nl = lane & 15, half = lane >> 4;
#pragma unroll
  for (int im = 0; im < 2; ++im)
#pragma unroll
    for (int in = 0; in < 4; ++in)
#pragma unroll
      for (int vg = 0; vg < 8; ++vg) {
        const int m = row0 + im * 16 + half * 8 + vg;
        const int n = col0 + in * 16 + nl;
        float v = acc[im][in][vg];
        if constexpr (HAS_BIAS) v += bias[n];
        if constexpr (RELU) v = fmaxf(v, 0.0f);
        const size_t idx = (size_t)m * N + n;
        if constexpr (HASADD) v += addF[idx];
        if constexpr (OUTF) outF[idx] = v;
        if constexpr (OUTB) outB[idx] = (__bf16)v;
        if constexpr (OUTTANH) outTanh[idx] = (__bf16)tanhf(v);
        if constexpr (OUTT) {  // transposed store: [batch][n][row-in-batch]
          const int bb = m / rpb;
          const int jr = m - bb * rpb;
          outT[(size_t)bb * ((size_t)N * rpb) + (size_t)n * rpb + jr] =
              (__bf16)v;
        }
      }
  (void)outF; (void)outB; (void)outTanh; (void)outT; (void)addF; (void)bias;
}

// ---------------------------------------------------------------------------
// scores = (q@k^T) * ((tanh(q)@tanh(k)^T + 1) * 0.5) / sqrt(DK)  (per batch)
// ---------------------------------------------------------------------------
__global__ __launch_bounds__(256) void k_scores(
    const bf16* __restrict__ qb, const bf16* __restrict__ tqb,
    const bf16* __restrict__ kb, const bf16* __restrict__ tkb,
    float* __restrict__ logits, int S, int DKd) {
  const int lane = threadIdx.x & 31;
  const int wave = threadIdx.x >> 5;
  const int wm = wave >> 2, wn = wave & 3;
  const int b = blockIdx.z;
  const int row0 = blockIdx.y * 64 + wm * 32;
  const int col0 = blockIdx.x * 128 + wn * 32;
  const size_t base = (size_t)b * S * DKd;
  const bf16* Aq = qb + base;
  const bf16* At = tqb + base;
  const bf16* Bk = kb + base;
  const bf16* Bh = tkb + base;

  const v8f vz = {};
  v8f sqk[2][2], sp[2][2];
  sqk[0][0]=vz; sqk[0][1]=vz; sqk[1][0]=vz; sqk[1][1]=vz;
  sp[0][0]=vz;  sp[0][1]=vz;  sp[1][0]=vz;  sp[1][1]=vz;

  for (int kk = 0; kk < DKd; kk += 32) {
    v16bf aq0 = load_a_bf16(Aq, DKd, row0, kk, lane);
    v16bf aq1 = load_a_bf16(Aq, DKd, row0 + 16, kk, lane);
    v16bf bk0 = load_b_bf16(Bk, DKd, col0, kk, lane);
    v16bf bk1 = load_b_bf16(Bk, DKd, col0 + 16, kk, lane);
    sqk[0][0] = wmma_bf16(aq0, bk0, sqk[0][0]);
    sqk[0][1] = wmma_bf16(aq0, bk1, sqk[0][1]);
    sqk[1][0] = wmma_bf16(aq1, bk0, sqk[1][0]);
    sqk[1][1] = wmma_bf16(aq1, bk1, sqk[1][1]);
    v16bf at0 = load_a_bf16(At, DKd, row0, kk, lane);
    v16bf at1 = load_a_bf16(At, DKd, row0 + 16, kk, lane);
    v16bf bt0 = load_b_bf16(Bh, DKd, col0, kk, lane);
    v16bf bt1 = load_b_bf16(Bh, DKd, col0 + 16, kk, lane);
    sp[0][0] = wmma_bf16(at0, bt0, sp[0][0]);
    sp[0][1] = wmma_bf16(at0, bt1, sp[0][1]);
    sp[1][0] = wmma_bf16(at1, bt0, sp[1][0]);
    sp[1][1] = wmma_bf16(at1, bt1, sp[1][1]);
  }

  const float scale = 0.044194173824159216f;  // 1/sqrt(512)
  const int nl = lane & 15, half = lane >> 4;
  float* out = logits + (size_t)b * S * S;
#pragma unroll
  for (int im = 0; im < 2; ++im)
#pragma unroll
    for (int in = 0; in < 2; ++in)
#pragma unroll
      for (int vg = 0; vg < 8; ++vg) {
        const int m = row0 + im * 16 + half * 8 + vg;
        const int n = col0 + in * 16 + nl;
        const float v =
            sqk[im][in][vg] * (sp[im][in][vg] + 1.0f) * 0.5f * scale;
        out[(size_t)m * S + n] = v;
      }
}

// ---------------------------------------------------------------------------
// Row softmax over S=2048, in-place f32 + bf16 copy. One block per row.
// ---------------------------------------------------------------------------
__global__ __launch_bounds__(256) void k_softmax(float* __restrict__ attn,
                                                 bf16* __restrict__ attnb,
                                                 int S) {
  const size_t row = blockIdx.x;
  float* p = attn + row * S;
  bf16* pb = attnb + row * S;
  const int tid = threadIdx.x, lane = tid & 31, wid = tid >> 5;
  __shared__ float ra[8], rb[8];
  float x[8];
  float mx = -3.4e38f;
#pragma unroll
  for (int i = 0; i < 8; ++i) { x[i] = p[tid + 256 * i]; mx = fmaxf(mx, x[i]); }
#pragma unroll
  for (int o = 16; o > 0; o >>= 1) mx = fmaxf(mx, __shfl_xor(mx, o, 32));
  if (lane == 0) ra[wid] = mx;
  __syncthreads();
  mx = ra[0];
#pragma unroll
  for (int w2 = 1; w2 < 8; ++w2) mx = fmaxf(mx, ra[w2]);
  float sum = 0.0f;
#pragma unroll
  for (int i = 0; i < 8; ++i) { x[i] = expf(x[i] - mx); sum += x[i]; }
#pragma unroll
  for (int o = 16; o > 0; o >>= 1) sum += __shfl_xor(sum, o, 32);
  if (lane == 0) rb[wid] = sum;
  __syncthreads();
  sum = 0.0f;
#pragma unroll
  for (int w2 = 0; w2 < 8; ++w2) sum += rb[w2];
  const float inv = 1.0f / sum;
#pragma unroll
  for (int i = 0; i < 8; ++i) {
    const float v = x[i] * inv;
    p[tid + 256 * i] = v;
    pb[tid + 256 * i] = (__bf16)v;
  }
}

// ---------------------------------------------------------------------------
// g = LN(relu(A0+A1)); e = LN(relu(A2)); geb = bf16(g*e). One block per row.
// ---------------------------------------------------------------------------
__global__ __launch_bounds__(256) void k_ln_ge(
    const float* __restrict__ A0, const float* __restrict__ A1,
    const float* __restrict__ A2, const float* __restrict__ gG,
    const float* __restrict__ bG, const float* __restrict__ gE,
    const float* __restrict__ bE, bf16* __restrict__ geb, int NF) {
  const size_t row = blockIdx.x;
  const int i = threadIdx.x, lane = i & 31, wid = i >> 5;
  const size_t idx = row * NF + i;
  const float h  = fmaxf(A0[idx] + A1[idx], 0.0f);
  const float er = fmaxf(A2[idx], 0.0f);
  float s0 = h, s1 = h * h, s2 = er, s3 = er * er;
#pragma unroll
  for (int o = 16; o > 0; o >>= 1) {
    s0 += __shfl_xor(s0, o, 32); s1 += __shfl_xor(s1, o, 32);
    s2 += __shfl_xor(s2, o, 32); s3 += __shfl_xor(s3, o, 32);
  }
  __shared__ float sm[4][8];
  if (lane == 0) { sm[0][wid]=s0; sm[1][wid]=s1; sm[2][wid]=s2; sm[3][wid]=s3; }
  __syncthreads();
  float S0=0, S1=0, S2=0, S3=0;
#pragma unroll
  for (int w2 = 0; w2 < 8; ++w2) {
    S0 += sm[0][w2]; S1 += sm[1][w2]; S2 += sm[2][w2]; S3 += sm[3][w2];
  }
  const float invN = 1.0f / NF;
  const float muH = S0 * invN, varH = S1 * invN - muH * muH;
  const float muE = S2 * invN, varE = S3 * invN - muE * muE;
  const float g = (h  - muH) * rsqrtf(varH + 1e-5f) * gG[i] + bG[i];
  const float e = (er - muE) * rsqrtf(varE + 1e-5f) * gE[i] + bE[i];
  geb[idx] = (__bf16)(g * e);
}

__global__ void k_cast(const float* __restrict__ src, bf16* __restrict__ dst,
                       int n) {
  const int i = blockIdx.x * blockDim.x + threadIdx.x;
  if (i < n) dst[i] = (__bf16)src[i];
}

// ---------------------------------------------------------------------------
extern "C" void kernel_launch(void* const* d_in, const int* in_sizes, int n_in,
                              void* d_out, int out_size, void* d_ws,
                              size_t ws_size, hipStream_t stream) {
  (void)in_sizes; (void)n_in; (void)out_size; (void)ws_size;
  const int B = 16, S = 2048, D = 512, DKc = 512, OC = 256;
  const int M = B * S;  // 32768 rows

  const float* x1 = (const float*)d_in[0];
  const float* x2 = (const float*)d_in[1];
  const float* Wq = (const float*)d_in[2];  const float* bq = (const float*)d_in[3];
  const float* Wk = (const float*)d_in[4];  const float* bk = (const float*)d_in[5];
  const float* Wv = (const float*)d_in[6];  const float* bv = (const float*)d_in[7];
  const float* W0 = (const float*)d_in[8];  const float* b0 = (const float*)d_in[9];
  const float* W1 = (const float*)d_in[10]; const float* b1 = (const float*)d_in[11];
  const float* W2 = (const float*)d_in[12]; const float* b2 = (const float*)d_in[13];
  const float* W3 = (const float*)d_in[14]; const float* b3 = (const float*)d_in[15];
  const float* gG = (const float*)d_in[16]; const float* betaG = (const float*)d_in[17];
  const float* gE = (const float*)d_in[18]; const float* betaE = (const float*)d_in[19];
  const float* Wc = (const float*)d_in[20]; const float* bc = (const float*)d_in[21];

  float* outMain = (float*)d_out;                    // [M, OC]
  float* attn    = (float*)d_out + (size_t)M * OC;   // [B, S, S]

  char* wsp = (char*)d_ws;
  auto carve = [&](size_t bytes) -> void* {
    void* p = (void*)wsp;
    wsp += (bytes + 255) & ~(size_t)255;
    return p;
  };
  const size_t nX = (size_t)M * D;
  bf16* qb    = (bf16*)carve(nX * 2);
  bf16* kbuf  = (bf16*)carve(nX * 2);
  bf16* tqb   = (bf16*)carve(nX * 2);
  bf16* tkb   = (bf16*)carve(nX * 2);
  bf16* vT    = (bf16*)carve(nX * 2);              // [B][D][S]
  bf16* qhb   = (bf16*)carve(nX * 2);
  bf16* attnb = (bf16*)carve((size_t)M * S * 2);
  float* A0f  = (float*)carve((size_t)M * OC * 4);
  float* A1f  = (float*)carve((size_t)M * OC * 4);
  float* A2f  = (float*)carve((size_t)M * OC * 4);
  float* A3f  = (float*)carve((size_t)M * OC * 4);
  bf16* geb   = (bf16*)carve((size_t)M * OC * 2);
  bf16* Wqb = (bf16*)carve((size_t)DKc * D * 2);
  bf16* Wkb = (bf16*)carve((size_t)DKc * D * 2);
  bf16* Wvb = (bf16*)carve((size_t)DKc * D * 2);
  bf16* W0b = (bf16*)carve((size_t)OC * DKc * 2);
  bf16* W1b = (bf16*)carve((size_t)OC * DKc * 2);
  bf16* W2b = (bf16*)carve((size_t)OC * DKc * 2);
  bf16* W3b = (bf16*)carve((size_t)OC * DKc * 2);
  bf16* Wcb = (bf16*)carve((size_t)OC * OC * 2);

  auto cast = [&](const float* s, bf16* dpt, int n) {
    k_cast<<<(n + 255) / 256, 256, 0, stream>>>(s, dpt, n);
  };
  cast(Wq, Wqb, DKc * D); cast(Wk, Wkb, DKc * D); cast(Wv, Wvb, DKc * D);
  cast(W0, W0b, OC * DKc); cast(W1, W1b, OC * DKc);
  cast(W2, W2b, OC * DKc); cast(W3, W3b, OC * DKc);
  cast(Wc, Wcb, OC * OC);

  const dim3 blk(256);
  auto grid_for = [](int M_, int N_) { return dim3((N_ + 255) / 256, (M_ + 63) / 64); };

  // Projections (A read in f32, cvt_pk_bf16 in-register).
  // q: store bf16 q + bf16 tanh(q)
  k_gemm<true, true, false, false, true, true, false, false>
      <<<grid_for(M, DKc), blk, 0, stream>>>(nullptr, x2, D, Wqb, D, 0, S, bq,
                                             M, DKc, D, nullptr, qb, tqb,
                                             nullptr, nullptr);
  // k: store bf16 k + bf16 tanh(k)
  k_gemm<true, true, false, false, true, true, false, false>
      <<<grid_for(M, DKc), blk, 0, stream>>>(nullptr, x1, D, Wkb, D, 0, S, bk,
                                             M, DKc, D, nullptr, kbuf, tkb,
                                             nullptr, nullptr);
  // v: store transposed bf16 v^T [B][D][S]
  k_gemm<true, true, false, false, false, false, true, false>
      <<<grid_for(M, DKc), blk, 0, stream>>>(nullptr, x1, D, Wvb, D, 0, S, bv,
                                             M, DKc, D, nullptr, nullptr,
                                             nullptr, vT, nullptr);

  // Attention logits -> d_out attention region, then softmax in place.
  dim3 gs(S / 128, S / 64, B);
  k_scores<<<gs, blk, 0, stream>>>(qb, tqb, kbuf, tkb, attn, S, DKc);
  k_softmax<<<M, blk, 0, stream>>>(attn, attnb, S);

  // q_hat = attn @ v   (Bt = v^T, per-batch stride)
  k_gemm<false, false, false, false, true, false, false, false>
      <<<grid_for(M, DKc), blk, 0, stream>>>(attnb, nullptr, S, vT, S,
                                             (long long)D * S, S, nullptr, M,
                                             DKc, S, nullptr, qhb, nullptr,
                                             nullptr, nullptr);

  // Four OUT linears (all relu'd, f32 out).
  k_gemm<false, true, true, true, false, false, false, false>
      <<<grid_for(M, OC), blk, 0, stream>>>(qb, nullptr, DKc, W0b, DKc, 0, S,
                                            b0, M, OC, DKc, A0f, nullptr,
                                            nullptr, nullptr, nullptr);
  k_gemm<false, true, true, true, false, false, false, false>
      <<<grid_for(M, OC), blk, 0, stream>>>(qhb, nullptr, DKc, W1b, DKc, 0, S,
                                            b1, M, OC, DKc, A1f, nullptr,
                                            nullptr, nullptr, nullptr);
  k_gemm<false, true, true, true, false, false, false, false>
      <<<grid_for(M, OC), blk, 0, stream>>>(qhb, nullptr, DKc, W2b, DKc, 0, S,
                                            b2, M, OC, DKc, A2f, nullptr,
                                            nullptr, nullptr, nullptr);
  k_gemm<false, true, true, true, false, false, false, false>
      <<<grid_for(M, OC), blk, 0, stream>>>(qb, nullptr, DKc, W3b, DKc, 0, S,
                                            b3, M, OC, DKc, A3f, nullptr,
                                            nullptr, nullptr, nullptr);

  // LayerNorms + gating product.
  k_ln_ge<<<M, blk, 0, stream>>>(A0f, A1f, A2f, gG, betaG, gE, betaE, geb, OC);

  // out = q_out + relu(ge @ Wc^T + bc)
  k_gemm<false, true, true, true, false, false, false, true>
      <<<grid_for(M, OC), blk, 0, stream>>>(geb, nullptr, OC, Wcb, OC, 0, S,
                                            bc, M, OC, OC, outMain, nullptr,
                                            nullptr, nullptr, A3f);
}
